// GraphQuantizer_31997506355311
// MI455X (gfx1250) — matrix operand order repaired and checked by reading, not compiled, not run on hardware
//
#include <hip/hip_runtime.h>
#include <hip/hip_bf16.h>
#include <math.h>

typedef __bf16 bf16x16 __attribute__((ext_vector_type(16)));
typedef __bf16 bf16x8  __attribute__((ext_vector_type(8)));
typedef float  f32x8   __attribute__((ext_vector_type(8)));

#define NB_GRAPHS 64

// ---------- helpers ----------
__device__ __forceinline__ __bf16 f2bf(float f) {
  unsigned u = __float_as_uint(f);
  unsigned r = u + 0x7FFFu + ((u >> 16) & 1u);   // round-to-nearest-even
  unsigned short h = (unsigned short)(r >> 16);
  __bf16 out;
  __builtin_memcpy(&out, &h, sizeof(out));
  return out;
}
__device__ __forceinline__ int f_ord(float f) {   // monotone float->int key
  int i = __float_as_int(f);
  return i >= 0 ? i : (i ^ 0x7fffffff);
}
__device__ __forceinline__ float ord_f(int i) {   // involution
  int j = i >= 0 ? i : (i ^ 0x7fffffff);
  return __int_as_float(j);
}
__device__ __forceinline__ float gelu_exact(float x) {
  return 0.5f * x * (1.0f + erff(x * 0.70710678118654752f));
}

// ---------- fills / converts ----------
__global__ void fill_f32(float* p, float v, size_t n) {
  size_t i = (size_t)blockIdx.x * blockDim.x + threadIdx.x;
  for (; i < n; i += (size_t)gridDim.x * blockDim.x) p[i] = v;
}
__global__ void fill_i32(int* p, int v, size_t n) {
  size_t i = (size_t)blockIdx.x * blockDim.x + threadIdx.x;
  for (; i < n; i += (size_t)gridDim.x * blockDim.x) p[i] = v;
}
__global__ void bias_bcast(float* out, const float* __restrict__ bias, int n, int m) {
  size_t total = (size_t)n * m;
  size_t i = (size_t)blockIdx.x * blockDim.x + threadIdx.x;
  for (; i < total; i += (size_t)gridDim.x * blockDim.x) out[i] = bias[i % m];
}
__global__ void cvt_bf16(const float* __restrict__ a, __bf16* __restrict__ b, size_t n) {
  size_t i = (size_t)blockIdx.x * blockDim.x + threadIdx.x;
  for (; i < n; i += (size_t)gridDim.x * blockDim.x) b[i] = f2bf(a[i]);
}
// pack W[K x M] (f32) into per-lane WMMA B-fragment order (bf16):
// PB[(((nt*(K/32))+kt)*32 + lane)*16 + j] = W[(kt*32 + (j>>3)*16 + (lane>>4)*8 + (j&7))*M + nt*16 + (lane&15)]
__global__ void pack_w(const float* __restrict__ W, __bf16* __restrict__ PB, int K, int M) {
  int total = K * M;
  int kt_n = K / 32;
  for (int i = blockIdx.x * blockDim.x + threadIdx.x; i < total;
       i += gridDim.x * blockDim.x) {
    int j = i & 15;
    int rest = i >> 4;
    int lane = rest & 31; rest >>= 5;
    int kt = rest % kt_n;
    int nt = rest / kt_n;
    int k = kt * 32 + (j >> 3) * 16 + (lane >> 4) * 8 + (j & 7);
    int m = nt * 16 + (lane & 15);
    PB[i] = f2bf(W[(size_t)k * M + m]);
  }
}

// ---------- layer 0: one-hot @ W == row gather ----------
__global__ void up_project(const int* __restrict__ colors, const float* __restrict__ w,
                           const float* __restrict__ b, float* __restrict__ out, int m) {
  int i = blockIdx.x;
  int j = threadIdx.x;
  out[(size_t)i * m + j] = w[(size_t)colors[i] * m + j] + b[j];
}

// ---------- WMMA GEMM: D = act(A[rows x K](bf16) @ PB(packed bf16) + bias) ----------
// one wave -> 16 rows x 64 cols (4 accumulators, 4 wmma per k-step), K-loop unrolled
template <int K>
__global__ void gemm_wmma(const __bf16* __restrict__ A, const __bf16* __restrict__ PB,
                          const float* __restrict__ bias, float* __restrict__ D,
                          int M, int act) {
  const int ntg   = blockIdx.x;       // group of 4 col tiles
  const int tileM = blockIdx.y;       // row tile
  const int lane  = threadIdx.x;      // 0..31
  const int half  = lane >> 4;
  const int l15   = lane & 15;
  constexpr int KT = K / 32;

  f32x8 acc0 = {}, acc1 = {}, acc2 = {}, acc3 = {};
  const __bf16* arow = A + (size_t)(tileM * 16 + l15) * K + half * 8;
  const __bf16* pb   = PB + ((size_t)ntg * 4 * KT) * 512 + lane * 16;

#pragma unroll
  for (int kt = 0; kt < KT; ++kt) {
    bf16x8 a0 = *(const bf16x8*)(arow + kt * 32);
    bf16x8 a1 = *(const bf16x8*)(arow + kt * 32 + 16);
    bf16x16 a = __builtin_shufflevector(a0, a1, 0, 1, 2, 3, 4, 5, 6, 7,
                                        8, 9, 10, 11, 12, 13, 14, 15);
#pragma unroll
    for (int nt = 0; nt < 4; ++nt) {
      const __bf16* pbn = pb + ((size_t)nt * KT + kt) * 512;
      bf16x8 b0 = *(const bf16x8*)(pbn);
      bf16x8 b1 = *(const bf16x8*)(pbn + 8);
      bf16x16 b = __builtin_shufflevector(b0, b1, 0, 1, 2, 3, 4, 5, 6, 7,
                                          8, 9, 10, 11, 12, 13, 14, 15);
      if (nt == 0) acc0 = __builtin_amdgcn_wmma_f32_16x16x32_bf16(false, a, false, b, (short)0, acc0, false, false);
      if (nt == 1) acc1 = __builtin_amdgcn_wmma_f32_16x16x32_bf16(false, a, false, b, (short)0, acc1, false, false);
      if (nt == 2) acc2 = __builtin_amdgcn_wmma_f32_16x16x32_bf16(false, a, false, b, (short)0, acc2, false, false);
      if (nt == 3) acc3 = __builtin_amdgcn_wmma_f32_16x16x32_bf16(false, a, false, b, (short)0, acc3, false, false);
    }
  }
#pragma unroll
  for (int nt = 0; nt < 4; ++nt) {
    f32x8 acc = nt == 0 ? acc0 : nt == 1 ? acc1 : nt == 2 ? acc2 : acc3;
    int col = (ntg * 4 + nt) * 16 + l15;
    float bv = bias ? bias[col] : 0.0f;
#pragma unroll
    for (int v = 0; v < 8; ++v) {
      int row = tileM * 16 + v + half * 8;
      float x = acc[v] + bv;
      if (act == 1) x = gelu_exact(x);
      D[(size_t)row * M + col] = x;
    }
  }
}

// ---------- GATv2 edge pipeline (C=128 per head), float4 vectorized ----------
__global__ void edge_logits_k(const int* __restrict__ src, const int* __restrict__ dst,
                              const float* __restrict__ el, const float* __restrict__ er,
                              const float* __restrict__ attn, float* __restrict__ logits,
                              int* __restrict__ nmax, int E, int H) {
  int wid  = (int)(((size_t)blockIdx.x * blockDim.x + threadIdx.x) >> 5);
  int lane = threadIdx.x & 31;
  if (wid >= E * H) return;
  int e = wid / H, h = wid - e * H;
  int s = src[e], d = dst[e];
  const float4* pel = (const float4*)(el + ((size_t)s * H + h) * 128);
  const float4* per = (const float4*)(er + ((size_t)d * H + h) * 128);
  const float4* pat = (const float4*)(attn + (size_t)h * 128);
  float4 va = pel[lane], vb = per[lane], vt = pat[lane];
  float x0 = va.x + vb.x, x1 = va.y + vb.y, x2 = va.z + vb.z, x3 = va.w + vb.w;
  x0 = x0 > 0.f ? x0 : 0.2f * x0;
  x1 = x1 > 0.f ? x1 : 0.2f * x1;
  x2 = x2 > 0.f ? x2 : 0.2f * x2;
  x3 = x3 > 0.f ? x3 : 0.2f * x3;
  float sum = x0 * vt.x + x1 * vt.y + x2 * vt.z + x3 * vt.w;
  for (int off = 16; off; off >>= 1) sum += __shfl_down(sum, off, 32);
  if (lane == 0) {
    logits[wid] = sum;
    atomicMax(&nmax[(size_t)d * H + h], f_ord(sum));
  }
}

__global__ void edge_expsum_k(const int* __restrict__ dst, const float* __restrict__ logits,
                              const int* __restrict__ nmax, float* __restrict__ ex,
                              float* __restrict__ denom, int E, int H) {
  int i = blockIdx.x * blockDim.x + threadIdx.x;
  if (i >= E * H) return;
  int e = i / H, h = i - e * H;
  int d = dst[e];
  float v = expf(logits[i] - ord_f(nmax[(size_t)d * H + h]));
  ex[i] = v;
  atomicAdd(&denom[(size_t)d * H + h], v);
}

__global__ void edge_agg_k(const int* __restrict__ src, const int* __restrict__ dst,
                           const float* __restrict__ el, const float* __restrict__ ex,
                           const float* __restrict__ denom, float* __restrict__ out,
                           int E, int H) {
  int wid  = (int)(((size_t)blockIdx.x * blockDim.x + threadIdx.x) >> 5);
  int lane = threadIdx.x & 31;
  if (wid >= E * H) return;
  int e = wid / H, h = wid - e * H;
  int s = src[e], d = dst[e];
  float alpha = ex[wid] / denom[(size_t)d * H + h];
  const float4* pel = (const float4*)(el + ((size_t)s * H + h) * 128);
  float* po = out + ((size_t)d * H + h) * 128;
  float4 v = pel[lane];
  atomicAdd(&po[lane * 4 + 0], alpha * v.x);
  atomicAdd(&po[lane * 4 + 1], alpha * v.y);
  atomicAdd(&po[lane * 4 + 2], alpha * v.z);
  atomicAdd(&po[lane * 4 + 3], alpha * v.w);
}

// ---------- layernorm over last dim (128), float4 ----------
__global__ void layernorm_k(const float* __restrict__ x, const float* __restrict__ g,
                            const float* __restrict__ b, float* __restrict__ y, int rows) {
  int wid  = (int)(((size_t)blockIdx.x * blockDim.x + threadIdx.x) >> 5);
  int lane = threadIdx.x & 31;
  if (wid >= rows) return;
  const float4* px = (const float4*)(x + (size_t)wid * 128);
  float4 v = px[lane];
  float s = v.x + v.y + v.z + v.w;
  for (int m = 16; m; m >>= 1) s += __shfl_xor(s, m, 32);
  float mu = s * (1.0f / 128.0f);
  float d0 = v.x - mu, d1 = v.y - mu, d2 = v.z - mu, d3 = v.w - mu;
  float q = d0 * d0 + d1 * d1 + d2 * d2 + d3 * d3;
  for (int m = 16; m; m >>= 1) q += __shfl_xor(q, m, 32);
  float rstd = rsqrtf(q * (1.0f / 128.0f) + 1e-5f);
  const float4* pg = (const float4*)g;
  const float4* pb = (const float4*)b;
  float4 gg = pg[lane], bb = pb[lane];
  float4 o;
  o.x = d0 * rstd * gg.x + bb.x;
  o.y = d1 * rstd * gg.y + bb.y;
  o.z = d2 * rstd * gg.z + bb.z;
  o.w = d3 * rstd * gg.w + bb.w;
  ((float4*)(y + (size_t)wid * 128))[lane] = o;
}

// ---------- global attention pooling ----------
__global__ void gate_k(const float* __restrict__ feats, const float* __restrict__ gw,
                       const float* __restrict__ gb, const int* __restrict__ seg,
                       float* __restrict__ gate, int* __restrict__ gmax, int n, int K) {
  int wid  = (int)(((size_t)blockIdx.x * blockDim.x + threadIdx.x) >> 5);
  int lane = threadIdx.x & 31;
  if (wid >= n) return;
  const float4* pf = (const float4*)(feats + (size_t)wid * K);
  const float4* pw = (const float4*)gw;
  float s = 0.f;
  for (int c = lane; c < K / 4; c += 32) {
    float4 a = pf[c], w = pw[c];
    s += a.x * w.x + a.y * w.y + a.z * w.z + a.w * w.w;
  }
  for (int m = 16; m; m >>= 1) s += __shfl_xor(s, m, 32);
  if (lane == 0) {
    float gv = s + gb[0];
    gate[wid] = gv;
    atomicMax(&gmax[seg[wid]], f_ord(gv));
  }
}
__global__ void gate_exp_k(const int* __restrict__ seg, const float* __restrict__ gate,
                           const int* __restrict__ gmax, float* __restrict__ ge,
                           float* __restrict__ gden, int n) {
  int i = blockIdx.x * blockDim.x + threadIdx.x;
  if (i >= n) return;
  float v = expf(gate[i] - ord_f(gmax[seg[i]]));
  ge[i] = v;
  atomicAdd(&gden[seg[i]], v);
}
__global__ void pool_k(const int* __restrict__ seg, const float* __restrict__ ge,
                       const float* __restrict__ gden, const float* __restrict__ feats,
                       float* __restrict__ pooled, int K) {
  int i = blockIdx.x;
  float w = ge[i] / gden[seg[i]];
  const float* pf = feats + (size_t)i * K;
  float* pp = pooled + (size_t)seg[i] * K;
  for (int c = threadIdx.x; c < K; c += blockDim.x) atomicAdd(&pp[c], w * pf[c]);
}
__global__ void tanh_out_k(const float* __restrict__ pooled, float* __restrict__ out, int n) {
  int i = blockIdx.x * blockDim.x + threadIdx.x;
  if (i < n) out[i] = tanhf(pooled[i]);
}

// ---------- host helpers ----------
static void gemm(hipStream_t s, const float* A, const float* W, const float* b, float* D,
                 int rows, int K, int M, int act, __bf16* Abf, __bf16* PB) {
  cvt_bf16<<<2048, 256, 0, s>>>(A, Abf, (size_t)rows * K);
  pack_w<<<256, 256, 0, s>>>(W, PB, K, M);
  dim3 g((unsigned)(M / 64), (unsigned)(rows / 16));
  if (K == 128) gemm_wmma<128><<<g, 32, 0, s>>>(Abf, PB, b, D, M, act);
  else          gemm_wmma<256><<<g, 32, 0, s>>>(Abf, PB, b, D, M, act);
}

static void run_gat(hipStream_t s, const int* src, const int* dst, int E, int N,
                    const float* el, const float* er, const float* attn, const float* bias,
                    int H, float* out, float* logits, float* exb, float* denom, int* nmax) {
  const int ORD_NEG_INF = (int)0x807FFFFF;   // f_ord(-inf)
  size_t nh = (size_t)N * H;
  fill_i32<<<256, 256, 0, s>>>(nmax, ORD_NEG_INF, nh);
  fill_f32<<<256, 256, 0, s>>>(denom, 0.0f, nh);
  bias_bcast<<<1024, 256, 0, s>>>(out, bias, N, H * 128);
  int totalEH = E * H;
  int wblocks = (totalEH + 7) / 8;
  edge_logits_k<<<wblocks, 256, 0, s>>>(src, dst, el, er, attn, logits, nmax, E, H);
  edge_expsum_k<<<(totalEH + 255) / 256, 256, 0, s>>>(dst, logits, nmax, exb, denom, E, H);
  edge_agg_k<<<wblocks, 256, 0, s>>>(src, dst, el, exb, denom, out, E, H);
}

// pytree (sorted-key) flatten order of setup_inputs()
enum {
  IN_DST = 0, IN_COLORS = 1,
  SEQ_ATTN = 2, SEQ_BIAS = 3, SEQ_DSTB = 4, SEQ_DSTW = 5, SEQ_SRCB = 6, SEQ_SRCW = 7,
  UP_ATTN = 8, UP_BIAS = 9, UP_DSTB = 10, UP_DSTW = 11, UP_SRCB = 12, UP_SRCW = 13,
  AT0 = 14, AT1 = 20,
  FFSEQ_LNB = 26, FFSEQ_LNG = 27, FFSEQ_W1B = 28, FFSEQ_W1W = 29, FFSEQ_W2B = 30, FFSEQ_W2W = 31,
  FF0 = 32, FF1 = 40,
  GATE_B = 48, GATE_W = 49,
  IN_SEG = 50, IN_SRC = 51
};

extern "C" void kernel_launch(void* const* d_in, const int* in_sizes, int n_in,
                              void* d_out, int out_size, void* d_ws, size_t ws_size,
                              hipStream_t stream) {
  const int* src    = (const int*)d_in[IN_SRC];
  const int* dst    = (const int*)d_in[IN_DST];
  const int* colors = (const int*)d_in[IN_COLORS];
  const int* seg    = (const int*)d_in[IN_SEG];
  const float* P[52];
  for (int i = 0; i < 52 && i < n_in; ++i) P[i] = (const float*)d_in[i];

  const int N = in_sizes[IN_COLORS];       // 20000
  const int E = in_sizes[IN_SRC];          // 180000

  // workspace layout (floats)
  float* feats  = (float*)d_ws;            // N*1024
  float* el     = feats  + (size_t)N * 1024;
  float* er     = el     + (size_t)N * 1024;
  float* gagg   = er     + (size_t)N * 1024;
  float* t2     = gagg   + (size_t)N * 1024;          // N*2048 (FF hidden)
  float* logits = t2     + (size_t)N * 2048;          // E*8
  float* exb    = logits + (size_t)E * 8;             // E*8
  float* denom  = exb    + (size_t)E * 8;             // N*8
  int*   nmax   = (int*)(denom + (size_t)N * 8);      // N*8
  float* gatev  = (float*)(nmax + (size_t)N * 8);     // N
  float* gev    = gatev  + N;                         // N
  float* gden   = gev    + N;                         // 64
  int*   gmax   = (int*)(gden + NB_GRAPHS);           // 64
  float* pooled = (float*)(gmax + NB_GRAPHS);         // 64*1024
  __bf16* Abf   = (__bf16*)(pooled + (size_t)NB_GRAPHS * 1024); // N*2048 bf16
  __bf16* PB    = Abf + (size_t)N * 2048;                       // 256K bf16 (weights)
  float* t1     = el;                                 // LN output reuses el

  // ---- layer 0: attn_up (one-hot -> row gather), H=1 ----
  up_project<<<N, 128, 0, stream>>>(colors, P[UP_SRCW], P[UP_SRCB], el, 128);
  up_project<<<N, 128, 0, stream>>>(colors, P[UP_DSTW], P[UP_DSTB], er, 128);
  run_gat(stream, src, dst, E, N, el, er, P[UP_ATTN], P[UP_BIAS], 1, feats,
          logits, exb, denom, nmax);

  // ---- depth layers: GAT(H=2) + FF(with w3) ----
  for (int d = 0; d < 2; ++d) {
    int A = (d == 0) ? AT0 : AT1;
    int F = (d == 0) ? FF0 : FF1;
    gemm(stream, feats, P[A + 5], P[A + 4], el, N, 128, 256, 0, Abf, PB);
    gemm(stream, feats, P[A + 3], P[A + 2], er, N, 128, 256, 0, Abf, PB);
    run_gat(stream, src, dst, E, N, el, er, P[A + 0], P[A + 1], 2, gagg,
            logits, exb, denom, nmax);
    layernorm_k<<<(N * 2 + 7) / 8, 256, 0, stream>>>(gagg, P[F + 1], P[F + 0], t1, N * 2);
    gemm(stream, t1, P[F + 3], P[F + 2], t2, N * 2, 128, 256, 1, Abf, PB);
    gemm(stream, t2, P[F + 5], P[F + 4], gagg, N * 2, 256, 128, 0, Abf, PB);
    gemm(stream, gagg, P[F + 7], P[F + 6], feats, N, 256, 128, 0, Abf, PB);
  }

  // ---- sequence layer: GAT(H=8) + FF(no w3) ----
  gemm(stream, feats, P[SEQ_SRCW], P[SEQ_SRCB], el, N, 128, 1024, 0, Abf, PB);
  gemm(stream, feats, P[SEQ_DSTW], P[SEQ_DSTB], er, N, 128, 1024, 0, Abf, PB);
  run_gat(stream, src, dst, E, N, el, er, P[SEQ_ATTN], P[SEQ_BIAS], 8, gagg,
          logits, exb, denom, nmax);
  layernorm_k<<<(N * 8 + 7) / 8, 256, 0, stream>>>(gagg, P[FFSEQ_LNG], P[FFSEQ_LNB], t1, N * 8);
  gemm(stream, t1, P[FFSEQ_W1W], P[FFSEQ_W1B], t2, N * 8, 128, 256, 1, Abf, PB);
  gemm(stream, t2, P[FFSEQ_W2W], P[FFSEQ_W2B], feats, N * 8, 256, 128, 0, Abf, PB);

  // ---- global attention pooling + tanh ----
  const int ORD_NEG_INF = (int)0x807FFFFF;
  fill_i32<<<1, 64, 0, stream>>>(gmax, ORD_NEG_INF, NB_GRAPHS);
  fill_f32<<<1, 64, 0, stream>>>(gden, 0.0f, NB_GRAPHS);
  fill_f32<<<256, 256, 0, stream>>>(pooled, 0.0f, (size_t)NB_GRAPHS * 1024);
  gate_k<<<(N + 7) / 8, 256, 0, stream>>>(feats, P[GATE_W], P[GATE_B], seg, gatev, gmax, N, 1024);
  gate_exp_k<<<(N + 255) / 256, 256, 0, stream>>>(seg, gatev, gmax, gev, gden, N);
  pool_k<<<N, 256, 0, stream>>>(seg, gev, gden, feats, pooled, 1024);
  tanh_out_k<<<(out_size + 255) / 256, 256, 0, stream>>>(pooled, (float*)d_out, out_size);
}
